// ConformerBlock_21749714387558
// MI455X (gfx1250) — matrix operand order repaired and checked by reading, not compiled, not run on hardware
//
#include <hip/hip_runtime.h>
#include <stdint.h>

// ---------------- types / helpers ----------------
typedef __attribute__((ext_vector_type(16))) __bf16 v16bf;
typedef __attribute__((ext_vector_type(8)))  float  v8f;

struct alignas(16) U4 { unsigned int a, b, c, d; };
union FragA { U4 q[2]; v16bf v; };            // 32B: one 16x32 A-frag or 32x16 B-frag (bf16)

static constexpr int Bc = 8, Sc = 1000, Dc = 512, Hc = 8, DKc = 64;
static constexpr int DIc = 2048, INNERc = 1024, KWc = 31, MAXLENc = 1000;
static constexpr int Mrows = Bc * Sc;          // 8000 token rows
static constexpr float EPSc = 1e-5f;

__device__ __forceinline__ unsigned short f2bf(float f) {
    unsigned int u = __float_as_uint(f);
    u += 0x7fffu + ((u >> 16) & 1u);           // round-to-nearest-even
    return (unsigned short)(u >> 16);
}
__device__ __forceinline__ float swish(float x) { return x / (1.f + __expf(-x)); }

// Low 32 bits of a generic pointer to LDS == LDS byte offset (aperture scheme).
__device__ __forceinline__ unsigned lds_off_u32(const void* p) {
    return (unsigned)(uintptr_t)p;
}
// CDNA5 async copy: 16B global -> LDS, tracked by ASYNCcnt (no VGPR data path).
__device__ __forceinline__ void async_g2l_b128(const void* gptr, unsigned lds_off) {
    asm volatile("global_load_async_to_lds_b128 %0, %1, off"
                 :: "v"(lds_off), "v"(gptr) : "memory");
}
__device__ __forceinline__ void wait_async0() {
    asm volatile("s_wait_asynccnt 0x0" ::: "memory");
}

// ---------------- f32 -> bf16 convert ----------------
__global__ void cvt_f32_bf16(const float* __restrict__ in, unsigned short* __restrict__ out, size_t n) {
    size_t i = (size_t)blockIdx.x * blockDim.x + threadIdx.x;
    size_t stride = (size_t)gridDim.x * blockDim.x;
    for (; i < n; i += stride) out[i] = f2bf(in[i]);
}

// ---------------- LayerNorm over D=512, one wave per row ----------------
template <int BFOUT>
__global__ __launch_bounds__(256) void ln_kernel(const float* __restrict__ x,
                                                 const float* __restrict__ g,
                                                 const float* __restrict__ b,
                                                 float* __restrict__ outF,
                                                 unsigned short* __restrict__ outBF,
                                                 int nrows) {
    int wave = threadIdx.x >> 5, lane = threadIdx.x & 31;
    int row = blockIdx.x * 8 + wave;
    if (row >= nrows) return;                  // no barriers in this kernel
    const float* xr = x + (size_t)row * Dc;
    float vals[16], s = 0.f, sq = 0.f;
    #pragma unroll
    for (int i = 0; i < 16; ++i) { float t = xr[lane + 32 * i]; vals[i] = t; s += t; sq += t * t; }
    #pragma unroll
    for (int m = 16; m >= 1; m >>= 1) { s += __shfl_xor(s, m, 32); sq += __shfl_xor(sq, m, 32); }
    float mean = s * (1.f / Dc);
    float var  = sq * (1.f / Dc) - mean * mean;
    float inv  = rsqrtf(var + EPSc);
    #pragma unroll
    for (int i = 0; i < 16; ++i) {
        int c = lane + 32 * i;
        float o = (vals[i] - mean) * inv * g[c] + b[c];
        if (BFOUT) outBF[(size_t)row * Dc + c] = f2bf(o);
        else       outF [(size_t)row * Dc + c] = o;
    }
}

// ---------------- bf16 WMMA GEMM: out = alpha*act(A @ W^T + bias) + residual --------------
// A: [M,K] bf16 row-major.  W: [N,K] bf16 row-major (weights stored [out,in]).
// Block tile 128x64, 8 waves (4Mx2N), 32x32 per wave. K multiple of 32, N multiple of 64.
// Staging: double-buffered GLOBAL_LOAD_ASYNC_TO_LDS_B128 (ASYNCcnt), overlapped with WMMA.
__global__ __launch_bounds__(256) void gemm_bf16_wmma(const unsigned short* __restrict__ A,
                                                      const unsigned short* __restrict__ W,
                                                      const float* __restrict__ bias,
                                                      const float* __restrict__ residual,
                                                      float* __restrict__ outF,
                                                      unsigned short* __restrict__ outBF,
                                                      int Mn, int Kn, int Nn,
                                                      float alpha, int act) {
    __shared__ alignas(16) unsigned short As[2][128 * 40];  // stride 40 halfs -> 16B-aligned chunks
    __shared__ alignas(16) unsigned short Bs[2][64 * 40];
    const int tid = threadIdx.x;
    const int wave = tid >> 5, lane = tid & 31;
    const int wm = wave & 3, wn = wave >> 2;
    const int l16 = lane & 15, lh = lane >> 4;
    const int m0 = blockIdx.x * 128, n0 = blockIdx.y * 64;

    v8f c[2][2] = {};

    const int aRow = tid >> 1, aKh = (tid & 1) * 16;   // A-stage: 2 threads/row, 32B each
    const int bRow = tid >> 2, bKh = (tid & 3) * 8;    // B-stage: 4 threads/row, 16B each
    int gA = m0 + aRow; if (gA > Mn - 1) gA = Mn - 1;  // clamp: OOB rows feed unstored C rows only
    const unsigned short* gpA = A + (size_t)gA * Kn + aKh;
    const unsigned short* gpB = W + (size_t)(n0 + bRow) * Kn + bKh;
    unsigned offA[2], offB[2];
    #pragma unroll
    for (int bsel = 0; bsel < 2; ++bsel) {
        offA[bsel] = lds_off_u32(&As[bsel][aRow * 40 + aKh]);
        offB[bsel] = lds_off_u32(&Bs[bsel][bRow * 40 + bKh]);
    }

    auto stage = [&](int bsel, int k0) {
        async_g2l_b128(gpA + k0,     offA[bsel]);
        async_g2l_b128(gpA + k0 + 8, offA[bsel] + 16);
        async_g2l_b128(gpB + k0,     offB[bsel]);
    };

    stage(0, 0);
    for (int k0 = 0; k0 < Kn; k0 += 32) {
        const int cur = (k0 >> 5) & 1;
        wait_async0();         // this wave's async loads into buf `cur` landed
        __syncthreads();       // all waves' loads landed; previous compute on buf cur^1 done
        if (k0 + 32 < Kn) stage(cur ^ 1, k0 + 32);

        // ---- fragments (contiguous k = 16*lh + s, same order in A and B) ----
        FragA a[2], b[2];
        #pragma unroll
        for (int mi = 0; mi < 2; ++mi) {
            int r = wm * 32 + mi * 16 + l16;
            a[mi].q[0] = *reinterpret_cast<const U4*>(&As[cur][r * 40 + (lh ? 16 : 0)]);      // K 0..7 / 16..23
            a[mi].q[1] = *reinterpret_cast<const U4*>(&As[cur][r * 40 + 8 + (lh ? 16 : 0)]);  // K 8..15 / 24..31
        }
        #pragma unroll
        for (int ni = 0; ni < 2; ++ni) {
            int n = wn * 32 + ni * 16 + l16;
            const U4* p = reinterpret_cast<const U4*>(&Bs[cur][n * 40 + (lh ? 16 : 0)]);      // K 0..15 / 16..31
            b[ni].q[0] = p[0]; b[ni].q[1] = p[1];
        }
        #pragma unroll
        for (int mi = 0; mi < 2; ++mi)
            #pragma unroll
            for (int ni = 0; ni < 2; ++ni)
                c[mi][ni] = __builtin_amdgcn_wmma_f32_16x16x32_bf16(
                    false, a[mi].v, false, b[ni].v, (short)0, c[mi][ni], false, false);
    }

    // ---- epilogue ----
    #pragma unroll
    for (int mi = 0; mi < 2; ++mi)
        #pragma unroll
        for (int ni = 0; ni < 2; ++ni) {
            int nn = n0 + wn * 32 + ni * 16 + l16;
            float bv = bias ? bias[nn] : 0.f;
            #pragma unroll
            for (int v = 0; v < 8; ++v) {
                int rr = m0 + wm * 32 + mi * 16 + v + 8 * lh;
                if (rr < Mn) {
                    float x = c[mi][ni][v] + bv;
                    if (act == 1) x = swish(x);
                    x *= alpha;
                    if (residual) x += residual[(size_t)rr * Nn + nn];
                    if (outF)  outF [(size_t)rr * Nn + nn] = x;
                    if (outBF) outBF[(size_t)rr * Nn + nn] = f2bf(x);
                }
            }
        }
}

// ---------------- fused attention (flash-style, rel-pos via skewed WMMA) ----------------
// q/k/v: [B,S,H,DK] bf16.  tbl: [2*MAXLEN-1, DK] bf16.  out: [B,S,D] f32 (pre-Wo).
// Block = 4 waves, 64 query rows. Each wave owns 16 rows, loops keys in steps of 32.
__global__ __launch_bounds__(128) void attn_wmma_kernel(const unsigned short* __restrict__ qbf,
                                                        const unsigned short* __restrict__ kbf,
                                                        const unsigned short* __restrict__ vbf,
                                                        const unsigned short* __restrict__ tbl,
                                                        float* __restrict__ outp) {
    __shared__ alignas(16) unsigned short Vs[64 * 40];     // V tile transposed: [dk][j], stride 40
    __shared__ alignas(16) float          Gs[4][16 * 48];  // per-wave rel-pos GEMM result
    __shared__ alignas(16) unsigned short Ps[4][16 * 40];  // per-wave P in A-frag transit layout

    const int tid = threadIdx.x, wave = tid >> 5, lane = tid & 31;
    const int l16 = lane & 15, lh = lane >> 4;
    const int b = blockIdx.z, h = blockIdx.y;
    const int i0 = blockIdx.x * 64 + wave * 16;
    const float scale = 0.125f;                // 1/sqrt(64)

    // ---- load Q fragments (16 rows x 64 features -> two 16x32 A-frags) ----
    FragA aq[2];
    {
        int i = i0 + l16; if (i > Sc - 1) i = Sc - 1;
        const unsigned short* qp = qbf + (((size_t)(b * Sc + i)) * Hc + h) * DKc;
        #pragma unroll
        for (int kt = 0; kt < 2; ++kt) {
            aq[kt].q[0] = *reinterpret_cast<const U4*>(qp + kt * 32 + (lh ? 16 : 0));
            aq[kt].q[1] = *reinterpret_cast<const U4*>(qp + kt * 32 + 8 + (lh ? 16 : 0));
        }
    }

    v8f o[4] = {};
    float mrun[8], lrun[8];
    #pragma unroll
    for (int v = 0; v < 8; ++v) { mrun[v] = -3.0e38f; lrun[v] = 0.f; }

    const int nsteps = (Sc + 31) / 32;
    for (int step = 0; step < nsteps; ++step) {
        const int j0 = step * 32;
        __syncthreads();
        // ---- cooperative V stage, transposed into LDS (transpose needs VGPR path) ----
        {
            int j = j0 + (tid & 31);
            int dk0 = (tid >> 5) * 16;
            union { U4 u[2]; unsigned short s[16]; } t;
            if (j < Sc) {
                const U4* p = reinterpret_cast<const U4*>(vbf + (((size_t)(b * Sc + j)) * Hc + h) * DKc + dk0);
                t.u[0] = p[0]; t.u[1] = p[1];
            } else {
                t.u[0] = U4{0,0,0,0}; t.u[1] = U4{0,0,0,0};
            }
            #pragma unroll
            for (int e = 0; e < 16; ++e) Vs[(dk0 + e) * 40 + (tid & 31)] = t.s[e];
        }
        __syncthreads();

        // ---- QK^T scores for two 16x16 j-tiles ----
        v8f sc[2];
        #pragma unroll
        for (int jt = 0; jt < 2; ++jt) {
            int j = j0 + jt * 16 + l16; if (j > Sc - 1) j = Sc - 1;
            const unsigned short* kp = kbf + (((size_t)(b * Sc + j)) * Hc + h) * DKc;
            FragA bk[2];
            #pragma unroll
            for (int kt = 0; kt < 2; ++kt) {
                const U4* p = reinterpret_cast<const U4*>(kp + kt * 32 + (lh ? 16 : 0));
                bk[kt].q[0] = p[0]; bk[kt].q[1] = p[1];
            }
            v8f z = {};
            z = __builtin_amdgcn_wmma_f32_16x16x32_bf16(false, aq[0].v, false, bk[0].v, (short)0, z, false, false);
            z = __builtin_amdgcn_wmma_f32_16x16x32_bf16(false, aq[1].v, false, bk[1].v, (short)0, z, false, false);
            sc[jt] = z;
        }

        // ---- rel-pos: G = Q @ tableSlice^T, 16x48 covering idx = base-31 .. base+15 ----
        {
            int idxmin = i0 - (j0 + 31) + (Sc - 1) + (MAXLENc - Sc);
            #pragma unroll
            for (int gt = 0; gt < 3; ++gt) {
                int trow = idxmin + gt * 16 + l16;
                if (trow < 0) trow = 0;
                if (trow > 2 * MAXLENc - 2) trow = 2 * MAXLENc - 2;
                const unsigned short* tp = tbl + (size_t)trow * DKc;
                FragA bt[2];
                #pragma unroll
                for (int kt = 0; kt < 2; ++kt) {
                    const U4* p = reinterpret_cast<const U4*>(tp + kt * 32 + (lh ? 16 : 0));
                    bt[kt].q[0] = p[0]; bt[kt].q[1] = p[1];
                }
                v8f g = {};
                g = __builtin_amdgcn_wmma_f32_16x16x32_bf16(false, aq[0].v, false, bt[0].v, (short)0, g, false, false);
                g = __builtin_amdgcn_wmma_f32_16x16x32_bf16(false, aq[1].v, false, bt[1].v, (short)0, g, false, false);
                #pragma unroll
                for (int v = 0; v < 8; ++v) Gs[wave][(v + 8 * lh) * 48 + gt * 16 + l16] = g[v];
            }
        }
        asm volatile("" ::: "memory");

        // ---- gather rel, mask, online softmax ----
        float pv[2][8], rowmax[8];
        #pragma unroll
        for (int v = 0; v < 8; ++v) {
            int mi = v + 8 * lh;
            float best = -3.0e38f;
            #pragma unroll
            for (int jt = 0; jt < 2; ++jt) {
                int mj = jt * 16 + l16;
                float rel = Gs[wave][mi * 48 + (31 + mi - mj)];
                float s = (sc[jt][v] + rel) * scale;
                if (j0 + mj >= Sc) s = -3.0e38f;
                pv[jt][v] = s;
                best = fmaxf(best, s);
            }
            rowmax[v] = best;
        }
        #pragma unroll
        for (int m = 1; m < 16; m <<= 1)
            #pragma unroll
            for (int v = 0; v < 8; ++v) rowmax[v] = fmaxf(rowmax[v], __shfl_xor(rowmax[v], m, 32));

        float corr[8], rowsum[8];
        #pragma unroll
        for (int v = 0; v < 8; ++v) {
            float newm = fmaxf(mrun[v], rowmax[v]);
            corr[v] = __expf(mrun[v] - newm);
            mrun[v] = newm;
            float s0 = __expf(pv[0][v] - newm);
            float s1 = __expf(pv[1][v] - newm);
            pv[0][v] = s0; pv[1][v] = s1;
            rowsum[v] = s0 + s1;
        }
        #pragma unroll
        for (int m = 1; m < 16; m <<= 1)
            #pragma unroll
            for (int v = 0; v < 8; ++v) rowsum[v] += __shfl_xor(rowsum[v], m, 32);
        #pragma unroll
        for (int v = 0; v < 8; ++v) {
            lrun[v] = lrun[v] * corr[v] + rowsum[v];
            #pragma unroll
            for (int dkt = 0; dkt < 4; ++dkt) o[dkt][v] *= corr[v];
        }

        // ---- P (C-layout) -> bf16 A-frag via LDS transit, then O += P @ V ----
        #pragma unroll
        for (int jt = 0; jt < 2; ++jt)
            #pragma unroll
            for (int v = 0; v < 8; ++v)
                Ps[wave][(v + 8 * lh) * 40 + jt * 16 + l16] = f2bf(pv[jt][v]);
        asm volatile("" ::: "memory");

        FragA ap;
        ap.q[0] = *reinterpret_cast<const U4*>(&Ps[wave][l16 * 40 + (lh ? 16 : 0)]);
        ap.q[1] = *reinterpret_cast<const U4*>(&Ps[wave][l16 * 40 + 8 + (lh ? 16 : 0)]);
        #pragma unroll
        for (int dkt = 0; dkt < 4; ++dkt) {
            FragA bv;
            const U4* p = reinterpret_cast<const U4*>(&Vs[(dkt * 16 + l16) * 40 + (lh ? 16 : 0)]);
            bv.q[0] = p[0]; bv.q[1] = p[1];
            o[dkt] = __builtin_amdgcn_wmma_f32_16x16x32_bf16(false, ap.v, false, bv.v, (short)0, o[dkt], false, false);
        }
    }

    // ---- normalize and store ----
    #pragma unroll
    for (int v = 0; v < 8; ++v) {
        int i = i0 + v + 8 * lh;
        if (i < Sc) {
            float invl = 1.f / lrun[v];
            #pragma unroll
            for (int dkt = 0; dkt < 4; ++dkt) {
                int dk = dkt * 16 + l16;
                outp[((size_t)(b * Sc + i)) * Dc + h * DKc + dk] = o[dkt][v] * invl;
            }
        }
    }
}

// ---------------- GLU: out = a * sigmoid(gate), h: [rows, 2*INNER] ----------------
__global__ void glu_kernel(const float* __restrict__ hbuf, float* __restrict__ out, int rows) {
    size_t total = (size_t)rows * INNERc;
    size_t idx = (size_t)blockIdx.x * blockDim.x + threadIdx.x;
    size_t stride = (size_t)gridDim.x * blockDim.x;
    for (; idx < total; idx += stride) {
        int c = (int)(idx % INNERc);
        size_t r = idx / INNERc;
        float a = hbuf[r * (2 * INNERc) + c];
        float g = hbuf[r * (2 * INNERc) + INNERc + c];
        out[idx] = a * (1.f / (1.f + __expf(-g)));
    }
}

// ---------------- depthwise conv (K=31, pad 15) + BN + swish, layout [B,S,INNER] --------
__global__ void dwconv_bn_swish_kernel(const float* __restrict__ in, const float* __restrict__ w,
                                       const float* __restrict__ dwb,
                                       const float* __restrict__ bng, const float* __restrict__ bnb,
                                       const float* __restrict__ bnm, const float* __restrict__ bnv,
                                       float* __restrict__ out) {
    size_t total = (size_t)Bc * Sc * INNERc;
    size_t idx = (size_t)blockIdx.x * blockDim.x + threadIdx.x;
    size_t stride = (size_t)gridDim.x * blockDim.x;
    for (; idx < total; idx += stride) {
        int c = (int)(idx % INNERc);
        size_t bs = idx / INNERc;
        int s = (int)(bs % Sc);
        int bb = (int)(bs / Sc);
        const float* base = in + ((size_t)bb * Sc) * INNERc + c;
        float acc = 0.f;
        #pragma unroll
        for (int t = 0; t < KWc; ++t) {
            int si = s + t - (KWc - 1) / 2;
            if (si >= 0 && si < Sc) acc += base[(size_t)si * INNERc] * w[c * KWc + t];
        }
        acc += dwb[c];
        acc = (acc - bnm[c]) * rsqrtf(bnv[c] + EPSc);
        acc = acc * bng[c] + bnb[c];
        out[idx] = swish(acc);
    }
}

// ---------------- host orchestration ----------------
extern "C" void kernel_launch(void* const* d_in, const int* in_sizes, int n_in,
                              void* d_out, int out_size, void* d_ws, size_t ws_size,
                              hipStream_t stream) {
    (void)in_sizes; (void)n_in; (void)out_size; (void)ws_size;

    const float* x0       = (const float*)d_in[0];
    const float* ffn1_g   = (const float*)d_in[1];
    const float* ffn1_bta = (const float*)d_in[2];
    const float* ffn1_w1  = (const float*)d_in[3];
    const float* ffn1_b1  = (const float*)d_in[4];
    const float* ffn1_w2  = (const float*)d_in[5];
    const float* ffn1_b2  = (const float*)d_in[6];
    const float* attn_g   = (const float*)d_in[7];
    const float* attn_bta = (const float*)d_in[8];
    const float* wq       = (const float*)d_in[9];
    const float* bq       = (const float*)d_in[10];
    const float* wk       = (const float*)d_in[11];
    const float* bk       = (const float*)d_in[12];
    const float* wv       = (const float*)d_in[13];
    const float* bv       = (const float*)d_in[14];
    const float* wo       = (const float*)d_in[15];
    const float* bo       = (const float*)d_in[16];
    const float* relpe    = (const float*)d_in[17];
    const float* conv_g   = (const float*)d_in[18];
    const float* conv_bta = (const float*)d_in[19];
    const float* pw1_w    = (const float*)d_in[20];
    const float* pw1_b    = (const float*)d_in[21];
    const float* dw_w     = (const float*)d_in[22];
    const float* dw_b     = (const float*)d_in[23];
    const float* bn_g     = (const float*)d_in[24];
    const float* bn_b     = (const float*)d_in[25];
    const float* bn_m     = (const float*)d_in[26];
    const float* bn_v     = (const float*)d_in[27];
    const float* pw2_w    = (const float*)d_in[28];
    const float* pw2_b    = (const float*)d_in[29];
    const float* ffn2_g   = (const float*)d_in[30];
    const float* ffn2_bta = (const float*)d_in[31];
    const float* ffn2_w1  = (const float*)d_in[32];
    const float* ffn2_b1  = (const float*)d_in[33];
    const float* ffn2_w2  = (const float*)d_in[34];
    const float* ffn2_b2  = (const float*)d_in[35];
    const float* fin_g    = (const float*)d_in[36];
    const float* fin_b    = (const float*)d_in[37];

    uint8_t* ws = (uint8_t*)d_ws;
    size_t cur = 0;
    auto alloc = [&](size_t bytes) -> void* {
        void* p = ws + cur;
        cur = (cur + bytes + 255) & ~(size_t)255;
        return p;
    };

    // bf16 weight pool
    unsigned short* wb_f1w1 = (unsigned short*)alloc((size_t)DIc * Dc * 2);
    unsigned short* wb_f1w2 = (unsigned short*)alloc((size_t)Dc * DIc * 2);
    unsigned short* wb_q    = (unsigned short*)alloc((size_t)Dc * Dc * 2);
    unsigned short* wb_k    = (unsigned short*)alloc((size_t)Dc * Dc * 2);
    unsigned short* wb_v    = (unsigned short*)alloc((size_t)Dc * Dc * 2);
    unsigned short* wb_o    = (unsigned short*)alloc((size_t)Dc * Dc * 2);
    unsigned short* wb_pw1  = (unsigned short*)alloc((size_t)2 * INNERc * Dc * 2);
    unsigned short* wb_pw2  = (unsigned short*)alloc((size_t)Dc * INNERc * 2);
    unsigned short* wb_f2w1 = (unsigned short*)alloc((size_t)DIc * Dc * 2);
    unsigned short* wb_f2w2 = (unsigned short*)alloc((size_t)Dc * DIc * 2);
    unsigned short* wb_tbl  = (unsigned short*)alloc((size_t)(2 * MAXLENc - 1) * DKc * 2);

    // activation buffers
    float*          F1   = (float*)alloc((size_t)Mrows * 2048 * 4);  // big f32 scratch
    float*          F2   = (float*)alloc((size_t)Mrows * 1024 * 4);  // GLU / conv scratch
    unsigned short* BF1  = (unsigned short*)alloc((size_t)Mrows * 2048 * 2);
    unsigned short* BFQ  = (unsigned short*)alloc((size_t)Mrows * Dc * 2);
    unsigned short* BFK  = (unsigned short*)alloc((size_t)Mrows * Dc * 2);
    unsigned short* BFV  = (unsigned short*)alloc((size_t)Mrows * Dc * 2);
    float*          res1 = (float*)alloc((size_t)Mrows * Dc * 4);
    float*          res2 = (float*)alloc((size_t)Mrows * Dc * 4);

    auto cvt = [&](const float* src, unsigned short* dst, size_t n) {
        unsigned blocks = (unsigned)((n + 2047) / 2048);
        cvt_f32_bf16<<<blocks, 256, 0, stream>>>(src, dst, n);
    };
    auto lnBF = [&](const float* xp, const float* g, const float* b, unsigned short* out) {
        ln_kernel<1><<<(Mrows + 7) / 8, 256, 0, stream>>>(xp, g, b, nullptr, out, Mrows);
    };
    auto gemm = [&](const unsigned short* A, const unsigned short* W, const float* bias,
                    const float* res, float* oF, unsigned short* oBF, int K, int N,
                    float alpha, int act) {
        dim3 grid((Mrows + 127) / 128, N / 64);
        gemm_bf16_wmma<<<grid, 256, 0, stream>>>(A, W, bias, res, oF, oBF, Mrows, K, N, alpha, act);
    };

    // 0) weights -> bf16
    cvt(ffn1_w1, wb_f1w1, (size_t)DIc * Dc);
    cvt(ffn1_w2, wb_f1w2, (size_t)Dc * DIc);
    cvt(wq, wb_q, (size_t)Dc * Dc);
    cvt(wk, wb_k, (size_t)Dc * Dc);
    cvt(wv, wb_v, (size_t)Dc * Dc);
    cvt(wo, wb_o, (size_t)Dc * Dc);
    cvt(pw1_w, wb_pw1, (size_t)2 * INNERc * Dc);
    cvt(pw2_w, wb_pw2, (size_t)Dc * INNERc);
    cvt(ffn2_w1, wb_f2w1, (size_t)DIc * Dc);
    cvt(ffn2_w2, wb_f2w2, (size_t)Dc * DIc);
    cvt(relpe, wb_tbl, (size_t)(2 * MAXLENc - 1) * DKc);

    // 1) FFN1 (macaron): x1 = x0 + 0.5*(swish(LN(x0)@w1^T+b1)@w2^T+b2)
    lnBF(x0, ffn1_g, ffn1_bta, BF1);
    gemm(BF1, wb_f1w1, ffn1_b1, nullptr, F1, nullptr, Dc, DIc, 1.f, 1);
    cvt(F1, BF1, (size_t)Mrows * DIc);
    gemm(BF1, wb_f1w2, ffn1_b2, x0, res1, nullptr, DIc, Dc, 0.5f, 0);

    // 2) Attention: x2 = x1 + (attn(LN(x1)) @ wo^T + bo)
    lnBF(res1, attn_g, attn_bta, BF1);
    gemm(BF1, wb_q, bq, nullptr, nullptr, BFQ, Dc, Dc, 1.f, 0);
    gemm(BF1, wb_k, bk, nullptr, nullptr, BFK, Dc, Dc, 1.f, 0);
    gemm(BF1, wb_v, bv, nullptr, nullptr, BFV, Dc, Dc, 1.f, 0);
    {
        dim3 grid((Sc + 63) / 64, Hc, Bc);
        attn_wmma_kernel<<<grid, 128, 0, stream>>>(BFQ, BFK, BFV, wb_tbl, F1);
    }
    cvt(F1, BF1, (size_t)Mrows * Dc);
    gemm(BF1, wb_o, bo, res1, res2, nullptr, Dc, Dc, 1.f, 0);

    // 3) Conv module: x3 = x2 + pw2(swish(BN(dwconv(GLU(pw1(LN(x2)))))))
    lnBF(res2, conv_g, conv_bta, BF1);
    gemm(BF1, wb_pw1, pw1_b, nullptr, F1, nullptr, Dc, 2 * INNERc, 1.f, 0);
    {
        size_t total = (size_t)Mrows * INNERc;
        unsigned blocks = (unsigned)((total + 255) / 256);
        glu_kernel<<<blocks, 256, 0, stream>>>(F1, F2, Mrows);
        dwconv_bn_swish_kernel<<<blocks, 256, 0, stream>>>(F2, dw_w, dw_b, bn_g, bn_b, bn_m, bn_v, F1);
    }
    cvt(F1, BF1, (size_t)Mrows * INNERc);
    gemm(BF1, wb_pw2, pw2_b, res2, res1, nullptr, INNERc, Dc, 1.f, 0);

    // 4) FFN2 (macaron): x4 = x3 + 0.5*FFN(x3)
    lnBF(res1, ffn2_g, ffn2_bta, BF1);
    gemm(BF1, wb_f2w1, ffn2_b1, nullptr, F1, nullptr, Dc, DIc, 1.f, 1);
    cvt(F1, BF1, (size_t)Mrows * DIc);
    gemm(BF1, wb_f2w2, ffn2_b2, res1, res2, nullptr, DIc, Dc, 0.5f, 0);

    // 5) final LN -> d_out (f32)
    ln_kernel<0><<<(Mrows + 7) / 8, 256, 0, stream>>>(res2, fin_g, fin_b, (float*)d_out, nullptr, Mrows);
}
